// Model_CR10_2_hsn_42898133353118
// MI455X (gfx1250) — matrix operand order repaired and checked by reading, not compiled, not run on hardware
//
#include <hip/hip_runtime.h>
#include <hip/hip_bf16.h>

typedef __attribute__((ext_vector_type(16))) __bf16 v16bf;
typedef __attribute__((ext_vector_type(8)))  __bf16 v8bf;
typedef __attribute__((ext_vector_type(8)))  float  v8f;

#define B_      4
#define H_      448
#define W_      512
#define H2_     224
#define W2_     256
#define CLS_    128
#define SLICES_ 8
#define HSTEP_  28
#define NEG_    0.01f
#define TX_     (W2_/16)   /* 16-pixel tiles per row (head, ds1) */
#define TXC_    (W2_/32)   /* 32-pixel tiles per row (conv)      */

__device__ __forceinline__ float lrelu(float v){ return v >= 0.f ? v : NEG_*v; }

__device__ __forceinline__ v16bf zero16(){
  v16bf v;
#pragma unroll
  for (int i=0;i<16;++i) v[i] = (__bf16)0.0f;
  return v;
}

__device__ __forceinline__ v16bf joinbf(v8bf lo, v8bf hi){
  return __builtin_shufflevector(lo, hi, 0,1,2,3,4,5,6,7,8,9,10,11,12,13,14,15);
}

// ds_swizzle SWAPX16 (group-of-32, xor_mask=0x10, and_mask=0x1f) -> 0x401F
__device__ __forceinline__ float swz16f(float v){
  return __int_as_float(__builtin_amdgcn_ds_swizzle(__float_as_int(v), 0x401F));
}
__device__ __forceinline__ int swz16i(int v){
  return __builtin_amdgcn_ds_swizzle(v, 0x401F);
}

// ---------------- weight packing: OIHW fp32 -> [cout][tap][ci] bf16 ----------------
__global__ void pack_w_kernel(const float* __restrict__ src, __bf16* __restrict__ dst,
                              int Cout, int Cin, int taps){
  int i = blockIdx.x*blockDim.x + threadIdx.x;
  int tot = Cout*Cin*taps;
  if (i >= tot) return;
  int ci = i % Cin; int r = i / Cin; int tap = r % taps; int co = r / taps;
  dst[i] = (__bf16)src[(size_t)(co*Cin + ci)*taps + tap];
}

// ---- ds1 weight packing: (32,16,5,5) fp32 -> [co][pair p<13][t<2][ci<16] bf16, zero-padded tap 25 ----
__global__ void pack_ds1_kernel(const float* __restrict__ src, __bf16* __restrict__ dst){
  int i = blockIdx.x*blockDim.x + threadIdx.x;    // 32*13*32 = 13312
  if (i >= 32*13*32) return;
  int ci = i & 15; int t = (i>>4)&1; int p = (i>>5)%13; int co = i/416;
  int tap = 2*p + t;
  float v = (tap < 25) ? src[(size_t)(co*16+ci)*25 + tap] : 0.f;
  dst[i] = (__bf16)v;
}

// ---------------- cs: 1 -> 16, 3x3, pad 1 (VALU; 0.26 GFLOP total) -> NHWC bf16 ----------------
__global__ void cs_kernel(const float* __restrict__ x, const float* __restrict__ w,
                          const float* __restrict__ bias, __bf16* __restrict__ out){
  int i = blockIdx.x*blockDim.x + threadIdx.x;
  if (i >= B_*H_*W_) return;
  int wc = i % W_; int r = i / W_; int h = r % H_; int b = r / H_;
  float v[9];
#pragma unroll
  for (int tap=0;tap<9;++tap){
    int dy = tap/3-1, dx = tap%3-1;
    int hy=h+dy, xx=wc+dx;
    v[tap] = (hy>=0 && hy<H_ && xx>=0 && xx<W_) ? x[((size_t)b*H_+hy)*W_+xx] : 0.f;
  }
#pragma unroll
  for (int co=0;co<16;++co){
    float a = bias[co];
#pragma unroll
    for (int tap=0;tap<9;++tap) a += w[co*9+tap]*v[tap];
    out[(size_t)i*16 + co] = (__bf16)lrelu(a);
  }
}

// ---------------- ds1: 16 -> 32, 5x5, stride 2, pad 2 via WMMA ----------------
// K chunks = tap-pair x 16 channels. Lanes 0-15: tap 2p (K 0-15); lanes 16-31: tap 2p+1 (K 16-31).
// One wave: 16 output pixels x 32 couts (2 M-tiles sharing each B fragment).
__global__ __launch_bounds__(32)
void ds1_wmma_kernel(const __bf16* __restrict__ in, const __bf16* __restrict__ wpk,
                     const float* __restrict__ bias, __bf16* __restrict__ out){
  int lane = threadIdx.x;
  int tile = blockIdx.x;              // B*H2*(W2/16)
  int wt = tile % TX_; int r = tile / TX_;
  int ho = r % H2_; int b = r / H2_;
  int n = lane & 15, hs = lane >> 4;
  int w0 = wt*16;
  v8f acc0 = {0.f,0.f,0.f,0.f,0.f,0.f,0.f,0.f};
  v8f acc1 = acc0;
  const int WK = 13*32;               // 416
  const __bf16* wr0 = wpk + (size_t)n*WK;
  const __bf16* wr1 = wpk + (size_t)(16+n)*WK;
#pragma unroll
  for (int p=0;p<13;++p){
    int tap = 2*p + hs;               // this half-wave's tap within the pair
    int ky = tap/5, kx = tap%5;
    int iy = 2*ho + ky - 2;
    int ix = 2*(w0+n) + kx - 2;
    bool valid = (tap < 25) && iy>=0 && iy<H_ && ix>=0 && ix<W_;
    v16bf bfrag = zero16();
    if (valid) bfrag = *(const v16bf*)(in + ((size_t)((size_t)b*H_+iy)*W_ + ix)*16);
    int kA = p*32 + hs*16;
    v16bf a0 = joinbf(*(const v8bf*)(wr0+kA), *(const v8bf*)(wr0+kA+8));
    v16bf a1 = joinbf(*(const v8bf*)(wr1+kA), *(const v8bf*)(wr1+kA+8));
    acc0 = __builtin_amdgcn_wmma_f32_16x16x32_bf16(false, a0, false, bfrag, (short)0, acc0, false, false);
    acc1 = __builtin_amdgcn_wmma_f32_16x16x32_bf16(false, a1, false, bfrag, (short)0, acc1, false, false);
  }
  size_t pix = ((size_t)b*H2_+ho)*W2_ + (w0+n);
  v8bf ob0, ob1;
#pragma unroll
  for (int g=0; g<8; ++g){
    int c0o = hs*8 + g;
    int c1o = 16 + hs*8 + g;
    ob0[g] = (__bf16)lrelu(acc0[g] + bias[c0o]);
    ob1[g] = (__bf16)lrelu(acc1[g] + bias[c1o]);
  }
  *(v8bf*)(out + pix*32 + hs*8)      = ob0;
  *(v8bf*)(out + pix*32 + 16 + hs*8) = ob1;
}

// ---------------- 3x3 conv, pad 1, implicit GEMM via bf16 WMMA, 2Mx2N register blocking ----------------
// One wave: 32 couts x 32 pixels => 4 accumulators; each k-step: 8 loads feed 4 WMMAs.
template<int CIN>
__global__ __launch_bounds__(32)
void conv3x3_wmma_kernel(const __bf16* __restrict__ in, const __bf16* __restrict__ wpk,
                         const float* __restrict__ bias, __bf16* __restrict__ out,
                         float* __restrict__ xlat, int Cout){
  int lane = threadIdx.x;
  int tile = blockIdx.x;              // B*H2*(W2/32)
  int wt = tile % TXC_; int r = tile / TXC_;
  int h = r % H2_; int b = r / H2_;
  int coutBase = blockIdx.y*32;
  int n = lane & 15, hs = lane >> 4;
  int w0 = wt*32;
  v8f acc[2][2];
  v8f z8 = {0.f,0.f,0.f,0.f,0.f,0.f,0.f,0.f};
  acc[0][0]=z8; acc[0][1]=z8; acc[1][0]=z8; acc[1][1]=z8;
  const int WK = 9*CIN;
  const __bf16* wr0 = wpk + (size_t)(coutBase+n)*WK;  // A row M = lane&15
  const __bf16* wr1 = wr0 + (size_t)16*WK;
#pragma unroll
  for (int tap=0;tap<9;++tap){
    int dy = tap/3 - 1, dx = tap%3 - 1;
    int hy = h+dy;
    int xx0 = w0+n+dx, xx1 = xx0+16;
    bool okY  = (hy>=0) && (hy<H2_);
    bool inb0 = okY && (xx0>=0) && (xx0<W2_);
    bool inb1 = okY && (xx1<W2_);     // xx1 >= 15 always
    const __bf16* rowb = in + ((size_t)((size_t)b*H2_+hy)*W2_)*CIN;
    if (inb0) __builtin_prefetch(rowb + (size_t)xx0*CIN, 0, 0);
    const __bf16* wt0 = wr0 + tap*CIN;
    const __bf16* wt1 = wr1 + tap*CIN;
#pragma unroll
    for (int c0=0;c0<CIN;c0+=32){
      int kA = c0 + hs*16;
      v16bf a0 = joinbf(*(const v8bf*)(wt0+kA), *(const v8bf*)(wt0+kA+8));
      v16bf a1 = joinbf(*(const v8bf*)(wt1+kA), *(const v8bf*)(wt1+kA+8));
      v16bf b0 = zero16(), b1 = zero16();
      if (inb0) b0 = *(const v16bf*)(rowb + (size_t)xx0*CIN + c0 + hs*16);
      if (inb1) b1 = *(const v16bf*)(rowb + (size_t)xx1*CIN + c0 + hs*16);
      acc[0][0] = __builtin_amdgcn_wmma_f32_16x16x32_bf16(false, a0, false, b0, (short)0, acc[0][0], false, false);
      acc[0][1] = __builtin_amdgcn_wmma_f32_16x16x32_bf16(false, a0, false, b1, (short)0, acc[0][1], false, false);
      acc[1][0] = __builtin_amdgcn_wmma_f32_16x16x32_bf16(false, a1, false, b0, (short)0, acc[1][0], false, false);
      acc[1][1] = __builtin_amdgcn_wmma_f32_16x16x32_bf16(false, a1, false, b1, (short)0, acc[1][1], false, false);
    }
  }
  // D: VGPR g -> M = 8*hs + g (within its 16-cout tile), N = lane&15
#pragma unroll
  for (int mi=0; mi<2; ++mi){
#pragma unroll
    for (int ni=0; ni<2; ++ni){
      int wcol = w0 + ni*16 + n;
      size_t pix = ((size_t)b*H2_+h)*W2_ + wcol;
      v8bf ob;
#pragma unroll
      for (int g=0; g<8; ++g){
        int co = coutBase + mi*16 + hs*8 + g;
        float v = lrelu(acc[mi][ni][g] + bias[co]);
        ob[g] = (__bf16)v;
        if (xlat) xlat[(((size_t)b*Cout + co)*H2_ + h)*W2_ + wcol] = v;
      }
      *(v8bf*)(out + pix*(size_t)Cout + coutBase + mi*16 + hs*8) = ob;
    }
  }
}

// ---------------- head: 128x128 classifier GEMM (WMMA) + softmax/argmax + reg/mask ----------------
__global__ __launch_bounds__(32)
void head_kernel(const __bf16* __restrict__ latent, const __bf16* __restrict__ wendc,
                 const float* __restrict__ endc_b, const float* __restrict__ endm_w,
                 const float* __restrict__ endm_b, const float* __restrict__ reg_w,
                 const float* __restrict__ reg_b, float* __restrict__ out_cls,
                 float* __restrict__ out_reg, float* __restrict__ out_msk){
  int lane = threadIdx.x;
  int tile = blockIdx.x;
  int wt = tile % TX_; int r = tile / TX_;
  int h = r % H2_; int b = r / H2_;
  int s = h / HSTEP_;
  int n = lane & 15, hs = lane >> 4;
  int w0 = wt*16;
  const __bf16* src = latent + ((size_t)((size_t)b*H2_+h)*W2_ + w0 + n)*CLS_;
  float L[64];
#pragma unroll
  for (int mt=0; mt<8; ++mt){
    v8f acc = {0.f,0.f,0.f,0.f,0.f,0.f,0.f,0.f};
    const __bf16* wrow = wendc + ((size_t)s*CLS_ + mt*16 + n)*CLS_;
#pragma unroll
    for (int c0=0;c0<CLS_;c0+=32){
      int kA = c0 + hs*16;
      v16bf afrag = joinbf(*(const v8bf*)(wrow+kA), *(const v8bf*)(wrow+kA+8));
      v16bf bfrag = *(const v16bf*)(src + c0 + hs*16);
      acc = __builtin_amdgcn_wmma_f32_16x16x32_bf16(false, afrag, false, bfrag,
                                                    (short)0, acc, false, false);
    }
#pragma unroll
    for (int g=0; g<8; ++g){
      int o = mt*16 + hs*8 + g;
      L[mt*8+g] = lrelu(acc[g] + endc_b[s*CLS_ + o]);
    }
  }
  // softmax over 128 classes: each lane holds 64; partner lane (lane ^ 16) holds the rest
  float mx = -3.4e38f;
#pragma unroll
  for (int i=0;i<64;++i) mx = fmaxf(mx, L[i]);
  mx = fmaxf(mx, swz16f(mx));
  float sum = 0.f;
#pragma unroll
  for (int i=0;i<64;++i){ L[i] = __expf(L[i]-mx); sum += L[i]; }
  sum += swz16f(sum);
  float inv = 1.f/sum;
  float bv = -1.f; int bo = CLS_;
  size_t pixBase = (size_t)b*CLS_*H2_*W2_ + (size_t)h*W2_ + (w0+n);
#pragma unroll
  for (int mt=0; mt<8; ++mt){
#pragma unroll
    for (int g=0; g<8; ++g){
      int o = mt*16 + hs*8 + g;
      float e = L[mt*8+g];
      out_cls[pixBase + (size_t)o*H2_*W2_] = e*inv;
      if (e > bv || (e == bv && o < bo)){ bv = e; bo = o; }
    }
  }
  float ov = swz16f(bv);
  int   oo = swz16i(bo);
  if (ov > bv || (ov == bv && oo < bo)){ bv = ov; bo = oo; }
  if (hs == 0){
    int full = bo + h*CLS_;
    const float* wg = reg_w + (size_t)full*CLS_;
    float rg = 0.f, mk = 0.f;
#pragma unroll 16
    for (int c=0;c<CLS_;++c){
      float lv = (float)src[c];
      rg += lv*wg[c];
      mk += lv*endm_w[c];
    }
    rg += reg_b[full];
    mk = lrelu(mk + endm_b[0]);
    size_t pp = ((size_t)b*H2_+h)*W2_ + (w0+n);
    out_reg[pp] = rg;
    out_msk[pp] = mk;
  }
}

extern "C" void kernel_launch(void* const* d_in, const int* in_sizes, int n_in,
                              void* d_out, int out_size, void* d_ws, size_t ws_size,
                              hipStream_t stream) {
  (void)in_sizes; (void)n_in; (void)out_size; (void)ws_size;
  const float* x     = (const float*)d_in[0];
  const float* cs_w  = (const float*)d_in[1];
  const float* cs_b  = (const float*)d_in[2];
  const float* ds1_w = (const float*)d_in[3];
  const float* ds1_b = (const float*)d_in[4];
  const float* c1_w = (const float*)d_in[5];  const float* c1_b = (const float*)d_in[6];
  const float* c2_w = (const float*)d_in[7];  const float* c2_b = (const float*)d_in[8];
  const float* c3_w = (const float*)d_in[9];  const float* c3_b = (const float*)d_in[10];
  const float* c4_w = (const float*)d_in[11]; const float* c4_b = (const float*)d_in[12];
  const float* c5_w = (const float*)d_in[13]; const float* c5_b = (const float*)d_in[14];
  const float* c6_w = (const float*)d_in[15]; const float* c6_b = (const float*)d_in[16];
  const float* endc_w = (const float*)d_in[17];
  const float* endc_b = (const float*)d_in[18];
  const float* endm_w = (const float*)d_in[19];
  const float* endm_b = (const float*)d_in[20];
  const float* reg_w  = (const float*)d_in[21];
  const float* reg_b  = (const float*)d_in[22];

  float* out = (float*)d_out;
  const size_t NPIX = (size_t)B_*H2_*W2_;                 // 229376
  float* out_cls  = out;                                   // (4,128,224,256)
  float* out_reg  = out + (size_t)CLS_*NPIX;               // (4,1,224,256)
  float* out_msk  = out_reg + NPIX;                        // (4,1,224,256)
  float* out_xlat = out_msk + NPIX;                        // (4,128,224,256)

  char* ws = (char*)d_ws;
  size_t off = 0;
  auto take = [&](size_t bytes)->char*{
    char* p = ws + off; off += (bytes + 255) & ~(size_t)255; return p;
  };
  __bf16* bx    = (__bf16*)take((size_t)29360128*2);  // ping (58.7 MB)
  __bf16* by    = (__bf16*)take((size_t)29360128*2);  // pong (58.7 MB)
  __bf16* wds1  = (__bf16*)take((size_t)13312*2);     // 32 x 13 pairs x 32
  __bf16* wc1   = (__bf16*)take((size_t)9216*2);
  __bf16* wc2   = (__bf16*)take((size_t)9216*2);
  __bf16* wc3   = (__bf16*)take((size_t)18432*2);
  __bf16* wc4   = (__bf16*)take((size_t)36864*2);
  __bf16* wc5   = (__bf16*)take((size_t)73728*2);
  __bf16* wc6   = (__bf16*)take((size_t)147456*2);
  __bf16* wendc = (__bf16*)take((size_t)131072*2);

  auto packLaunch = [&](const float* src, __bf16* dst, int Cout, int Cin, int taps){
    int tot = Cout*Cin*taps;
    pack_w_kernel<<<(tot+255)/256, 256, 0, stream>>>(src, dst, Cout, Cin, taps);
  };
  pack_ds1_kernel<<<(13312+255)/256, 256, 0, stream>>>(ds1_w, wds1);
  packLaunch(c1_w,  wc1,  32, 32, 9);
  packLaunch(c2_w,  wc2,  32, 32, 9);
  packLaunch(c3_w,  wc3,  64, 32, 9);
  packLaunch(c4_w,  wc4,  64, 64, 9);
  packLaunch(c5_w,  wc5, 128, 64, 9);
  packLaunch(c6_w,  wc6, 128,128, 9);
  packLaunch(endc_w, wendc, SLICES_*CLS_, CLS_, 1);   // identity repack to bf16

  cs_kernel<<<(B_*H_*W_+255)/256, 256, 0, stream>>>(x, cs_w, cs_b, bx);          // a0 -> bx
  int ntiles16 = B_*H2_*TX_;    // 14336
  ds1_wmma_kernel<<<ntiles16, 32, 0, stream>>>(bx, wds1, ds1_b, by);             // a1 -> by

  int ntiles32 = B_*H2_*TXC_;   // 7168
  conv3x3_wmma_kernel<32><<<dim3(ntiles32,1), 32, 0, stream>>>(by, wc1, c1_b, bx, nullptr, 32);
  conv3x3_wmma_kernel<32><<<dim3(ntiles32,1), 32, 0, stream>>>(bx, wc2, c2_b, by, nullptr, 32);
  conv3x3_wmma_kernel<32><<<dim3(ntiles32,2), 32, 0, stream>>>(by, wc3, c3_b, bx, nullptr, 64);
  conv3x3_wmma_kernel<64><<<dim3(ntiles32,2), 32, 0, stream>>>(bx, wc4, c4_b, by, nullptr, 64);
  conv3x3_wmma_kernel<64><<<dim3(ntiles32,4), 32, 0, stream>>>(by, wc5, c5_b, bx, nullptr, 128);
  conv3x3_wmma_kernel<128><<<dim3(ntiles32,4), 32, 0, stream>>>(bx, wc6, c6_b, by, out_xlat, 128);

  head_kernel<<<ntiles16, 32, 0, stream>>>(by, wendc, endc_b, endm_w, endm_b,
                                           reg_w, reg_b, out_cls, out_reg, out_msk);
}
